// Network_div_78374563217914
// MI455X (gfx1250) — matrix operand order repaired and checked by reading, not compiled, not run on hardware
//
#include <hip/hip_runtime.h>
#include <hip/hip_bf16.h>
#include <math.h>

// ---------------------------------------------------------------------------
// Reference: per-node 9x9 embedded-Gaussian non-local block + 7x7 box-count
// gate over a 1280x1280 binary mask. Bandwidth-bound (~55 MB total traffic).
// Counts image is computed as two banded matmuls per 16x16 tile using
// v_wmma_f32_16x16x32_f16 (exact: 0/1 data, sums <= 49, fp16-representable).
// ---------------------------------------------------------------------------

constexpr int IMG  = 1280;   // H == W
constexpr int FEAT = 9;

typedef __attribute__((ext_vector_type(16))) _Float16 v16h;
typedef __attribute__((ext_vector_type(8)))  float    v8f;

// ------------------------------ Kernel 1 -----------------------------------
// One wave (32 threads) per 16x16 tile of the counts image.
// Stage 1: H[k][j] = sum_{dc=-3..3} gt[r0-3+k][c0+j+dc]  (k = 0..21)
//          computed as  A(gt rows, 16x32 f16) x Band1(32x16)  -- two WMMAs
//          to cover the 22-row halo (rows r0-3 .. r0+18).
// Stage 2: counts[r0+i][c0+j] = sum_{k=i..i+6} H[k][j]
//          computed as  Band2(16x32) x H(32x16)             -- one WMMA.
// Band matrices are zero for k >= 22, so halo garbage never contributes.
// ---------------------------------------------------------------------------
__global__ __launch_bounds__(32)
void counts_wmma_kernel(const int* __restrict__ gt, float* __restrict__ counts) {
  __shared__ _Float16 Hs[32][16];

  const int lane  = threadIdx.x;       // 0..31
  const int laneM = lane & 15;         // M (A-layout) / N (B,D-layout)
  const int hi    = lane >> 4;         // lane group
  const int r0    = blockIdx.y * 16;
  const int c0    = blockIdx.x * 16;

  // Per-lane K indices of the A-matrix 16x32 f16 layout (ISA 7.12.2):
  // lane<16:  V0..3 -> K=0..7,  V4..7 -> K=16..23
  // lane>=16: V0..3 -> K=8..15, V4..7 -> K=24..31
  int K[16];
#pragma unroll
  for (int v = 0; v < 8; ++v) {
    int klo = (hi == 0) ? ((v < 4) ? 2 * v : 2 * v + 8)
                        : ((v < 4) ? 2 * v + 8 : 2 * v + 16);
    K[2 * v]     = klo;
    K[2 * v + 1] = klo + 1;
  }

  // B-layout (32x16, 8 VGPRs): col N = lane%16; element e holds K = K[e], as in A
  // (AMD table order); kbase below is not used by b1 or b2.
  const int kbase = hi ? 16 : 0;

  // Band1[k][j] = 1 iff j <= k <= j+6  (j = output column within tile)
  v16h b1;
#pragma unroll
  for (int e = 0; e < 16; ++e) {
    int k = K[e];
    b1[e] = (laneM <= k && k <= laneM + 6) ? (_Float16)1.0f : (_Float16)0.0f;
  }

  // A operands: gt data with zero padding outside the image.
  v16h a1, a2;
#pragma unroll
  for (int e = 0; e < 16; ++e) {
    const int c  = c0 - 3 + K[e];
    const int r1 = r0 - 3 + laneM;     // rows r0-3 .. r0+12  (k = m)
    const int r2 = r0 + 13 + laneM;    // rows r0+13 .. r0+28 (k = 16+m; k>=22 unused)
    const bool cok = (c >= 0) & (c < IMG);
    float g1 = (cok && r1 >= 0 && r1 < IMG) ? (float)gt[r1 * IMG + c] : 0.0f;
    float g2 = (cok && r2 >= 0 && r2 < IMG) ? (float)gt[r2 * IMG + c] : 0.0f;
    a1[e] = (_Float16)g1;
    a2[e] = (_Float16)g2;
  }

  v8f h1 = {}, h2 = {};
  h1 = __builtin_amdgcn_wmma_f32_16x16x32_f16(false, a1, false, b1, (short)0, h1, false, false);
  h2 = __builtin_amdgcn_wmma_f32_16x16x32_f16(false, a2, false, b1, (short)0, h2, false, false);

  // C/D layout -> LDS [k][j]:  lane L, VGPR v  is  (M = v + 8*(L>=16), N = L%16)
#pragma unroll
  for (int v = 0; v < 8; ++v) {
    const int m = v + 8 * hi;
    Hs[m][laneM]      = (_Float16)h1[v];   // k = m
    Hs[16 + m][laneM] = (_Float16)h2[v];   // k = 16 + m
  }
  __syncthreads();

  // Band2[i][k] = 1 iff i <= k <= i+6  (i = M = laneM in A-layout)
  v16h aband;
#pragma unroll
  for (int e = 0; e < 16; ++e) {
    int k = K[e];
    aband[e] = (laneM <= k && k <= laneM + 6) ? (_Float16)1.0f : (_Float16)0.0f;
  }
  // Re-load H from LDS in B-layout.
  v16h b2;
#pragma unroll
  for (int e = 0; e < 16; ++e)
    b2[e] = Hs[K[e]][laneM];

  v8f acc = {};
  acc = __builtin_amdgcn_wmma_f32_16x16x32_f16(false, aband, false, b2, (short)0, acc, false, false);

#pragma unroll
  for (int v = 0; v < 8; ++v) {
    const int m = v + 8 * hi;
    counts[(r0 + m) * IMG + (c0 + laneM)] = acc[v];
  }
}

// ------------------------------ Kernel 2 -----------------------------------
// One thread per node: gather cnt, 9x9 softmax attention, gated residual.
// ---------------------------------------------------------------------------
template <bool USE_COUNTS>
__global__ __launch_bounds__(256)
void nonlocal_kernel(const float* __restrict__ feat,
                     const int*   __restrict__ idx,
                     const int*   __restrict__ gt,
                     const float* __restrict__ params,
                     const float* __restrict__ counts,
                     float* __restrict__ out, int n) {
  const int t = blockIdx.x * blockDim.x + threadIdx.x;
  if (t >= n) return;

  const int r = idx[2 * t];
  const int c = idx[2 * t + 1];

  float cnt;
  if (USE_COUNTS) {
    cnt = counts[r * IMG + c];
  } else {
    int s = 0;
#pragma unroll
    for (int dr = -3; dr <= 3; ++dr)
#pragma unroll
      for (int dc = -3; dc <= 3; ++dc) {
        const int rr = r + dr, cc = c + dc;
        if (rr >= 0 && rr < IMG && cc >= 0 && cc < IMG) s += gt[rr * IMG + cc];
      }
    cnt = (float)s;
  }
  const bool take = (cnt >= 2.0f);

  const float wg = params[0], bg = params[1];
  const float wt = params[2], bt = params[3];
  const float wp = params[4], bp = params[5];
  const float ww = params[6], bw = params[7];

  float x[FEAT], th[FEAT], ph[FEAT], g[FEAT];
#pragma unroll
  for (int j = 0; j < FEAT; ++j) {
    x[j]  = feat[t * FEAT + j];
    th[j] = wt * x[j] + bt;
    ph[j] = wp * x[j] + bp;
    g[j]  = wg * x[j] + bg;
  }

#pragma unroll
  for (int i = 0; i < FEAT; ++i) {
    float p[FEAT];
    float m = -INFINITY;
#pragma unroll
    for (int j = 0; j < FEAT; ++j) {
      p[j] = th[i] * ph[j];
      m = fmaxf(m, p[j]);
    }
    float s = 0.0f, ys = 0.0f;
#pragma unroll
    for (int j = 0; j < FEAT; ++j) {
      const float e = __expf(p[j] - m);   // v_exp_f32
      s  += e;
      ys += e * g[j];
    }
    const float z = ww * (ys / s) + bw + x[i];
    out[t * FEAT + i] = take ? z : x[i];
  }
}

// ------------------------------ Launcher -----------------------------------
extern "C" void kernel_launch(void* const* d_in, const int* in_sizes, int n_in,
                              void* d_out, int out_size, void* d_ws, size_t ws_size,
                              hipStream_t stream) {
  const float* feat   = (const float*)d_in[0];   // (n, 9) float32
  const int*   idx    = (const int*)  d_in[1];   // (n, 2) int32
  // d_in[2] == indices (duplicate), d_in[3] == cube_size (==7)
  const int*   gt     = (const int*)  d_in[4];   // (1280, 1280) int32
  const float* params = (const float*)d_in[5];   // (8,) float32

  const int n = in_sizes[0] / FEAT;
  float* out = (float*)d_out;
  float* counts = (float*)d_ws;

  const size_t counts_bytes = (size_t)IMG * IMG * sizeof(float);
  const int blocks = (n + 255) / 256;

  if (ws_size >= counts_bytes) {
    dim3 grid(IMG / 16, IMG / 16);
    counts_wmma_kernel<<<grid, 32, 0, stream>>>(gt, counts);
    nonlocal_kernel<true><<<blocks, 256, 0, stream>>>(feat, idx, gt, params, counts, out, n);
  } else {
    nonlocal_kernel<false><<<blocks, 256, 0, stream>>>(feat, idx, gt, params, counts, out, n);
  }
}
